// Net_72086731096527
// MI455X (gfx1250) — compile-verified, matrix operand order checked by reading, not run on hardware
//
#include <hip/hip_runtime.h>
#include <hip/hip_bf16.h>
#include <math.h>

typedef __attribute__((ext_vector_type(16))) _Float16 v16h;
typedef __attribute__((ext_vector_type(8)))  float    v8f;

// ---------------------------------------------------------------------------
// WMMA fragment builders (v_wmma_f32_16x16x32_f16, wave32).
// A (16x32, f16): lane m=lane&15 (both halves), base=8*(lane>=16),
//   elements 0..7  = K k0+base+0..7,  elements 8..15 = K k0+16+base+0..7.
// B (32x16, f16): lane holds column n0+(lane&15), same K mapping as A.
// C (16x16, f32): VGPR v -> row v + 8*(lane>=16), col = lane&15.
// ---------------------------------------------------------------------------
__device__ __forceinline__ v16h frag_a_f32(const float* __restrict__ A, long ld,
                                           int k0, int lane) {
  const int half = (lane >> 4) & 1;
  const int m    = lane & 15;
  const float* r = A + (long)m * ld + k0 + half * 8;
  const float4 c0 = *(const float4*)(r + 0);
  const float4 c1 = *(const float4*)(r + 4);
  const float4 c2 = *(const float4*)(r + 16);
  const float4 c3 = *(const float4*)(r + 20);
  v16h f;
  f[0]=(_Float16)c0.x;  f[1]=(_Float16)c0.y;  f[2]=(_Float16)c0.z;  f[3]=(_Float16)c0.w;
  f[4]=(_Float16)c1.x;  f[5]=(_Float16)c1.y;  f[6]=(_Float16)c1.z;  f[7]=(_Float16)c1.w;
  f[8]=(_Float16)c2.x;  f[9]=(_Float16)c2.y;  f[10]=(_Float16)c2.z; f[11]=(_Float16)c2.w;
  f[12]=(_Float16)c3.x; f[13]=(_Float16)c3.y; f[14]=(_Float16)c3.z; f[15]=(_Float16)c3.w;
  return f;
}

__device__ __forceinline__ v16h frag_b_f32(const float* __restrict__ Bw, int ldn,
                                           int k0, int n0, int lane) {
  const int half = (lane >> 4) & 1;
  const int n    = n0 + (lane & 15);
  const float* p = Bw + (long)(k0 + half * 8) * ldn + n;
  const float* q = p + (long)16 * ldn;
  v16h f;
#pragma unroll
  for (int e = 0; e < 8; ++e) f[e]     = (_Float16)p[(long)e * ldn];
#pragma unroll
  for (int e = 0; e < 8; ++e) f[8 + e] = (_Float16)q[(long)e * ldn];
  return f;
}

// ---------------------------------------------------------------------------
// Scaled Laplacian, scratch-free.
// Pass 1: one thread per node: deg_i = sum of weights of out-edges of i;
//         dis[node] = rsqrt(deg); zero the diagonal entry L[g][i][i].
// Pass 2: one thread per edge: L[g][d][s] = -dis[s]*w*dis[d]  (every
//         off-diagonal slot has exactly one edge: fully-connected graphs).
// ---------------------------------------------------------------------------
__global__ void node_dis(const int* __restrict__ es, const float* __restrict__ ew,
                         float* __restrict__ L, float* __restrict__ dis, long Nn) {
  long n = (long)blockIdx.x * blockDim.x + threadIdx.x;
  if (n >= Nn) return;
  const long g = n >> 3;
  const long base = g * 56;
  float deg = 0.f;
  for (int e = 0; e < 56; ++e) {
    if ((long)es[base + e] == n) deg += ew[base + e];
  }
  dis[n] = deg > 0.f ? rsqrtf(deg) : 0.f;
  const int i = (int)(n & 7);
  L[g * 64 + i * 8 + i] = 0.f;     // zero diagonal
}

__global__ void edge_L(const int* __restrict__ es, const int* __restrict__ ed,
                       const float* __restrict__ ew, const float* __restrict__ dis,
                       float* __restrict__ L, long E) {
  long e = (long)blockIdx.x * blockDim.x + threadIdx.x;
  if (e >= E) return;
  const int s = es[e], d = ed[e];
  const long g = (long)(s >> 3);
  const int ls = s & 7, ld = d & 7;
  L[g * 64 + ld * 8 + ls] = -dis[s] * ew[e] * dis[d];
}

// ---------------------------------------------------------------------------
// Fused ChebConv layer: one wave owns 2 graphs (16 rows).
//  Phase 0: wave 0 stages all pre-swizzled f16 W fragments in LDS (shared by
//           the whole block -> weight reads amortized 8x).
//  Phase 1: load L (2x64) and X (16xFIN) to LDS.
//  Phase 2: recurrence Tx1 = L x, Tx2 = 2 L Tx1 - x, Tx3 = 2 L Tx2 - Tx1,
//           concatenated in LDS as T[16][4*FIN]  (K = kcheb*FIN + f, matching
//           row-major flatten of W (K,FIN,FOUT) -> (4*FIN, FOUT)).
//  Phase 3: WMMA: Y(16xFOUT) = relu(T @ Wcat + b); A-fragments hoisted out of
//           the N-tile loop.
// Block = 256 threads = 8 waves = 16 graphs.  Grid = B/16.
// ---------------------------------------------------------------------------
template <int FIN, int FOUT>
__global__ void __launch_bounds__(256)
cheb_fused(const float* __restrict__ X, const float* __restrict__ L,
           const float* __restrict__ W, const float* __restrict__ bias,
           float* __restrict__ Y) {
  constexpr int KTOT = 4 * FIN;
  constexpr int NT = FOUT / 16;
  constexpr int NK = KTOT / 32;
  __shared__ __align__(16) float Ls[8][128];
  __shared__ __align__(16) float T[8][16][KTOT];
  __shared__ __align__(32) _Float16 Wf[NT][NK][32][16];

  const int wave = threadIdx.x >> 5;
  const int lane = threadIdx.x & 31;
  const int g0   = (blockIdx.x * 8 + wave) * 2;   // first graph of this pair

  // Phase 0/1: cooperative loads; wave 0 additionally stages W fragments
  for (int i = lane; i < 128; i += 32) Ls[wave][i] = L[(long)g0 * 64 + i];
  for (int i = lane; i < 16 * FIN; i += 32) {
    int r = i / FIN, f = i - r * FIN;
    T[wave][r][f] = X[(long)(g0 * 8) * FIN + i];
  }
  if (wave == 0) {
#pragma unroll
    for (int nt = 0; nt < NT; ++nt)
#pragma unroll
      for (int kt = 0; kt < NK; ++kt)
        *(v16h*)&Wf[nt][kt][lane][0] = frag_b_f32(W, FOUT, kt * 32, nt * 16, lane);
  }
  __syncthreads();

  // Phase 2: column-parallel Chebyshev recurrence
  {
    const int gl = lane >> 4;       // graph within pair
    const int c  = lane & 15;
    for (int f = c; f < FIN; f += 16) {
      float t0[8], t1[8], t2[8];
#pragma unroll
      for (int n = 0; n < 8; ++n) t0[n] = T[wave][gl * 8 + n][f];
#pragma unroll
      for (int d = 0; d < 8; ++d) {
        float s = 0.f;
#pragma unroll
        for (int k = 0; k < 8; ++k) s += Ls[wave][gl * 64 + d * 8 + k] * t0[k];
        t1[d] = s;
        T[wave][gl * 8 + d][FIN + f] = s;
      }
#pragma unroll
      for (int d = 0; d < 8; ++d) {
        float s = 0.f;
#pragma unroll
        for (int k = 0; k < 8; ++k) s += Ls[wave][gl * 64 + d * 8 + k] * t1[k];
        t2[d] = 2.f * s - t0[d];
        T[wave][gl * 8 + d][2 * FIN + f] = t2[d];
      }
#pragma unroll
      for (int d = 0; d < 8; ++d) {
        float s = 0.f;
#pragma unroll
        for (int k = 0; k < 8; ++k) s += Ls[wave][gl * 64 + d * 8 + k] * t2[k];
        T[wave][gl * 8 + d][3 * FIN + f] = 2.f * s - t1[d];
      }
    }
  }
  __syncthreads();

  // Phase 3: WMMA channel mixing (uniform control flow, EXEC all ones)
  const float* Abase = &T[wave][0][0];
  v16h afr[NK];
#pragma unroll
  for (int kt = 0; kt < NK; ++kt) afr[kt] = frag_a_f32(Abase, KTOT, kt * 32, lane);

  const int half = lane >> 4, nl = lane & 15;
#pragma unroll
  for (int nt = 0; nt < NT; ++nt) {
    v8f acc = {};
#pragma unroll
    for (int kt = 0; kt < NK; ++kt) {
      v16h b = *(const v16h*)&Wf[nt][kt][lane][0];
      acc = __builtin_amdgcn_wmma_f32_16x16x32_f16(false, afr[kt], false, b,
                                                   (short)0, acc, false, false);
    }
    const int col = nt * 16 + nl;
    const float bv = bias[col];
#pragma unroll
    for (int v = 0; v < 8; ++v) {
      const int row = g0 * 8 + v + half * 8;
      float val = acc[v] + bv;
      val = val > 0.f ? val : 0.f;
      Y[(long)row * FOUT + col] = val;
    }
  }
}

// ---------------------------------------------------------------------------
// WMMA GEMM: C(MxN) = relu(A(MxK,f32,row-major) @ W(KxN,f32) + bias).
// Block = 8 waves sharing one 16-col N-tile; wave 0 stages the pre-swizzled
// f16 B panel for all K in LDS; each wave computes 4 M-tiles (64 rows), so
// one staged B fragment feeds 4 WMMAs.  K <= 512, K%32==0, N%16==0.
// Grid.x = (N/16) * ceil(M/512).
// ---------------------------------------------------------------------------
__global__ void __launch_bounds__(256)
gemm_bias_relu(const float* __restrict__ A, const float* __restrict__ W,
               const float* __restrict__ bias, float* __restrict__ C,
               int M, int N, int K) {
  __shared__ __align__(32) _Float16 Bf[16][32][16];   // [kstep][lane][elem]
  const int wave   = threadIdx.x >> 5;
  const int lane   = threadIdx.x & 31;
  const int tilesN = N >> 4;
  const int tn     = blockIdx.x % tilesN;
  const int gmBase = (blockIdx.x / tilesN) * 8;
  const int nK     = K >> 5;

  if (wave == 0) {
    for (int kt = 0; kt < nK; ++kt)
      *(v16h*)&Bf[kt][lane][0] = frag_b_f32(W, N, kt * 32, tn * 16, lane);
  }
  __syncthreads();

  const long row0 = (long)(gmBase + wave) * 64;
  if (row0 >= M) return;                       // wave-uniform exit (post-barrier)

  v8f acc0 = {}, acc1 = {}, acc2 = {}, acc3 = {};
  const float* A0 = A + row0 * K;
  for (int kt = 0; kt < nK; ++kt) {
    const int k0 = kt * 32;
    v16h b  = *(const v16h*)&Bf[kt][lane][0];
    v16h a0 = frag_a_f32(A0,              K, k0, lane);
    v16h a1 = frag_a_f32(A0 + 16L * K,    K, k0, lane);
    v16h a2 = frag_a_f32(A0 + 32L * K,    K, k0, lane);
    v16h a3 = frag_a_f32(A0 + 48L * K,    K, k0, lane);
    acc0 = __builtin_amdgcn_wmma_f32_16x16x32_f16(false, a0, false, b, (short)0, acc0, false, false);
    acc1 = __builtin_amdgcn_wmma_f32_16x16x32_f16(false, a1, false, b, (short)0, acc1, false, false);
    acc2 = __builtin_amdgcn_wmma_f32_16x16x32_f16(false, a2, false, b, (short)0, acc2, false, false);
    acc3 = __builtin_amdgcn_wmma_f32_16x16x32_f16(false, a3, false, b, (short)0, acc3, false, false);
  }

  const int half = lane >> 4;
  const int col  = tn * 16 + (lane & 15);
  const float bv = bias[col];
  v8f accs[4] = {acc0, acc1, acc2, acc3};
#pragma unroll
  for (int t = 0; t < 4; ++t) {
#pragma unroll
    for (int v = 0; v < 8; ++v) {
      const long row = row0 + t * 16 + v + half * 8;
      float val = accs[t][v] + bv;
      val = val > 0.f ? val : 0.f;
      C[row * N + col] = val;
    }
  }
}

// ---------------------------------------------------------------------------
// Classifier: logits = H(128) @ W(128x5) + b; out = log_softmax(logits).
// One thread per graph; W stays resident in L2.
// ---------------------------------------------------------------------------
__global__ void fc3_logsoftmax(const float* __restrict__ H, const float* __restrict__ W,
                               const float* __restrict__ b, float* __restrict__ out, int B) {
  int g = blockIdx.x * blockDim.x + threadIdx.x;
  if (g >= B) return;
  float logit[5];
#pragma unroll
  for (int j = 0; j < 5; ++j) logit[j] = b[j];
  const float* h = H + (long)g * 128;
  for (int k = 0; k < 128; ++k) {
    const float hv = h[k];
#pragma unroll
    for (int j = 0; j < 5; ++j) logit[j] += hv * W[k * 5 + j];
  }
  float m = logit[0];
#pragma unroll
  for (int j = 1; j < 5; ++j) m = fmaxf(m, logit[j]);
  float s = 0.f;
#pragma unroll
  for (int j = 0; j < 5; ++j) s += __expf(logit[j] - m);
  const float lse = m + __logf(s);
#pragma unroll
  for (int j = 0; j < 5; ++j) out[(long)g * 5 + j] = logit[j] - lse;
}

// ---------------------------------------------------------------------------
extern "C" void kernel_launch(void* const* d_in, const int* in_sizes, int n_in,
                              void* d_out, int out_size, void* d_ws, size_t ws_size,
                              hipStream_t stream) {
  const float* x    = (const float*)d_in[0];
  const int*   es   = (const int*)d_in[1];
  const int*   ed   = (const int*)d_in[2];
  const float* ew   = (const float*)d_in[3];
  const float* w1   = (const float*)d_in[4];
  const float* b1   = (const float*)d_in[5];
  const float* w2   = (const float*)d_in[6];
  const float* b2   = (const float*)d_in[7];
  const float* w3   = (const float*)d_in[8];
  const float* b3   = (const float*)d_in[9];
  const float* fc1w = (const float*)d_in[10];
  const float* fc1b = (const float*)d_in[11];
  const float* fc2w = (const float*)d_in[12];
  const float* fc2b = (const float*)d_in[13];
  const float* fc3w = (const float*)d_in[14];
  const float* fc3b = (const float*)d_in[15];
  float* out = (float*)d_out;

  const long N = in_sizes[0] / 8;   // nodes  (262144)
  const long B = N / 8;             // graphs (32768)
  const long E = in_sizes[1];       // edges  (B*56)

  // Workspace layout (floats), ~127 MB total with overlays:
  //   Lw : B*64 | Y1 : N*16 | Y2 : N*32 | Y3 : N*64 | dis : N
  //   H1 (B*256 = N*32) overlays [Y1,Y2+N*16) after layer 3 consumed them.
  //   H2 (B*128 = N*16) overlays Y3 after FC1 consumed it.
  float* ws  = (float*)d_ws;
  float* Lw  = ws;
  float* Y1  = Lw + B * 64;
  float* Y2  = Y1 + N * 16;
  float* Y3  = Y2 + N * 32;
  float* dis = Y3 + N * 64;
  float* H1  = Y1;
  float* H2  = Y3;

  node_dis<<<(int)((N + 255) / 256), 256, 0, stream>>>(es, ew, Lw, dis, N);
  edge_L<<<(int)((E + 255) / 256), 256, 0, stream>>>(es, ed, ew, dis, Lw, E);

  cheb_fused<8, 16><<<(int)(B / 16), 256, 0, stream>>>(x,  Lw, w1, b1, Y1);
  cheb_fused<16, 32><<<(int)(B / 16), 256, 0, stream>>>(Y1, Lw, w2, b2, Y2);
  cheb_fused<32, 64><<<(int)(B / 16), 256, 0, stream>>>(Y2, Lw, w3, b3, Y3);

  {  // FC1: (B x 512) @ (512 x 256)   — Y3 (N x 64) viewed row-major as (B x 512)
    int blocks = (256 / 16) * (int)((B + 511) / 512);
    gemm_bias_relu<<<blocks, 256, 0, stream>>>(Y3, fc1w, fc1b, H1, (int)B, 256, 512);
  }
  {  // FC2: (B x 256) @ (256 x 128)
    int blocks = (128 / 16) * (int)((B + 511) / 512);
    gemm_bias_relu<<<blocks, 256, 0, stream>>>(H1, fc2w, fc2b, H2, (int)B, 128, 256);
  }
  fc3_logsoftmax<<<(int)((B + 255) / 256), 256, 0, stream>>>(H2, fc3w, fc3b, out, (int)B);
}
